// LatentDirectionEncoder_38878043963452
// MI455X (gfx1250) — matrix operand order repaired and checked by reading, not compiled, not run on hardware
//
#include <hip/hip_runtime.h>
#include <math.h>

typedef _Float16 f16;
typedef __attribute__((ext_vector_type(16))) _Float16 v16h;
typedef __attribute__((ext_vector_type(8)))  float    v8f;

__device__ inline float siluf(float x)     { return x / (1.f + expf(-x)); }
__device__ inline float softplusf(float x) { return x > 20.f ? x : log1pf(expf(x)); }

__device__ inline float wave_sum(float v) {
#pragma unroll
  for (int o = 16; o > 0; o >>= 1) v += __shfl_xor(v, o, 32);
  return v;
}

// low 32 bits of a generic pointer to a __shared__ object == LDS byte address
__device__ inline unsigned lds_addr_of(const void* p) {
  return (unsigned)(unsigned long long)p;
}

// ---------------------------------------------------------------------------
// WMMA GEMM:  C[M,N] = A[M,K] * B[N,K]^T   (B stored row-major N x K)
// aColMajor: A[m,k] read as A[k*M+m]   outColMajor: C written as C[n*M+m]
// Block: 256 thr = 8 waves (2x4); tile 64(M) x 128(N); each wave owns 32x32
// (2x2 frags -> 4 v_wmma per 32-wide K step).
// Interior blocks: double-buffered global_load_async_to_lds_b32 (ASYNCcnt) so
// tile k+1 DMA overlaps tile k WMMA; one s_wait_asynccnt + one barrier per step.
// Edge blocks: branchless clamped loads (no exec-mask branches).
// Requires K % 32 == 0 (true for every call site).
// ---------------------------------------------------------------------------
__global__ void __launch_bounds__(256) gemm_wmma_kernel(
    const float* __restrict__ A, const float* __restrict__ B,
    const float* __restrict__ bias, float* __restrict__ C,
    int M, int N, int K, int aColMajor, int outColMajor)
{
  __shared__ float Asf[2][64][33];
  __shared__ float Bsf[2][128][33];
  const int tid  = threadIdx.x;
  const int lane = tid & 31;
  const int wave = tid >> 5;
  const int wm = wave & 1;   // 0..1 -> 32-row slice
  const int wn = wave >> 1;  // 0..3 -> 32-col slice
  const int row0 = blockIdx.y * 64;
  const int col0 = blockIdx.x * 128;
  const int m16 = lane & 15, hsel = lane >> 4;
  const size_t aRS = aColMajor ? (size_t)1 : (size_t)K;  // stride along m
  const size_t aCS = aColMajor ? (size_t)M : (size_t)1;  // stride along k
  const bool full = (row0 + 64 <= M) && (col0 + 128 <= N);
  const int nk = K >> 5;
  v8f acc00 = {}, acc01 = {}, acc10 = {}, acc11 = {};

  auto issue_tile = [&](int kt, int buf) {  // async DMA tile -> LDS (raw f32)
    int k0 = kt << 5;
#pragma unroll
    for (int e = 0; e < 8; e++) {
      int i = tid + e * 256;
      int r = i >> 5, kk = i & 31;
      const float* gp = A + (size_t)(row0 + r) * aRS + (size_t)(k0 + kk) * aCS;
      unsigned lo = lds_addr_of(&Asf[buf][r][kk]);
      asm volatile("global_load_async_to_lds_b32 %0, %1, off"
                   :: "v"(lo), "v"(gp) : "memory");
    }
#pragma unroll
    for (int e = 0; e < 16; e++) {
      int i = tid + e * 256;
      int r = i >> 5, kk = i & 31;
      const float* gp = B + (size_t)(col0 + r) * K + (size_t)(k0 + kk);
      unsigned lo = lds_addr_of(&Bsf[buf][r][kk]);
      asm volatile("global_load_async_to_lds_b32 %0, %1, off"
                   :: "v"(lo), "v"(gp) : "memory");
    }
  };

  auto compute_tile = [&](int buf) {
    v16h a0, a1, b0, b1;
#pragma unroll
    for (int i = 0; i < 8; i++) {
      int ka = ((i < 4) ? 0 : 16) + hsel * 8 + 2 * (i & 3);
      a0[2 * i]     = (f16)Asf[buf][wm * 32 + m16][ka];
      a0[2 * i + 1] = (f16)Asf[buf][wm * 32 + m16][ka + 1];
      a1[2 * i]     = (f16)Asf[buf][wm * 32 + 16 + m16][ka];
      a1[2 * i + 1] = (f16)Asf[buf][wm * 32 + 16 + m16][ka + 1];
      int kb = ((i < 4) ? 0 : 16) + hsel * 8 + 2 * (i & 3);
      b0[2 * i]     = (f16)Bsf[buf][wn * 32 + m16][kb];
      b0[2 * i + 1] = (f16)Bsf[buf][wn * 32 + m16][kb + 1];
      b1[2 * i]     = (f16)Bsf[buf][wn * 32 + 16 + m16][kb];
      b1[2 * i + 1] = (f16)Bsf[buf][wn * 32 + 16 + m16][kb + 1];
    }
    acc00 = __builtin_amdgcn_wmma_f32_16x16x32_f16(false, a0, false, b0, (short)0, acc00, false, false);
    acc01 = __builtin_amdgcn_wmma_f32_16x16x32_f16(false, a0, false, b1, (short)0, acc01, false, false);
    acc10 = __builtin_amdgcn_wmma_f32_16x16x32_f16(false, a1, false, b0, (short)0, acc10, false, false);
    acc11 = __builtin_amdgcn_wmma_f32_16x16x32_f16(false, a1, false, b1, (short)0, acc11, false, false);
  };

  if (full) {
    issue_tile(0, 0);
    for (int kt = 0; kt < nk; kt++) {
      asm volatile("s_wait_asynccnt 0x0" ::: "memory");  // this wave's tile kt landed
      __syncthreads();  // all waves' tile kt landed; all waves done with tile kt-1
      if (kt + 1 < nk) issue_tile(kt + 1, (kt + 1) & 1);  // DMA overlaps compute below
      compute_tile(kt & 1);
    }
  } else {
    for (int kt = 0; kt < nk; kt++) {
      int k0 = kt << 5;
      __syncthreads();
      // branchless clamped loads (K%32==0 so gk always valid)
#pragma unroll
      for (int e = 0; e < 8; e++) {
        int i = tid + e * 256;
        int r = i >> 5, kk = i & 31;
        int gm = row0 + r, gk = k0 + kk;
        int cm = gm < M ? gm : M - 1;
        float v = A[(size_t)cm * aRS + (size_t)gk * aCS];
        Asf[0][r][kk] = (gm < M) ? v : 0.f;
      }
#pragma unroll
      for (int e = 0; e < 16; e++) {
        int i = tid + e * 256;
        int r = i >> 5, kk = i & 31;
        int gn = col0 + r, gk = k0 + kk;
        int cn = gn < N ? gn : N - 1;
        float v = B[(size_t)cn * K + (size_t)gk];
        Bsf[0][r][kk] = (gn < N) ? v : 0.f;
      }
      __syncthreads();
      compute_tile(0);
    }
  }

  auto store_tile = [&](v8f acc, int mbase, int nbase) {
#pragma unroll
    for (int r = 0; r < 8; r++) {
      int gm = mbase + hsel * 8 + r;
      int gn = nbase + m16;
      if (gm < M && gn < N) {
        float v = acc[r];
        if (bias) v += bias[gn];
        if (outColMajor) C[(size_t)gn * M + gm] = v;
        else             C[(size_t)gm * N + gn] = v;
      }
    }
  };
  store_tile(acc00, row0 + wm * 32,      col0 + wn * 32);
  store_tile(acc01, row0 + wm * 32,      col0 + wn * 32 + 16);
  store_tile(acc10, row0 + wm * 32 + 16, col0 + wn * 32);
  store_tile(acc11, row0 + wm * 32 + 16, col0 + wn * 32 + 16);
}

// ---------------------------------------------------------------------------
// im2col for 3x3 pad-1 convs: col[m][k], m=y*W+x, k=(ci*3+ky)*3+kx
// ---------------------------------------------------------------------------
__global__ void im2col3x3_kernel(const float* in, float* col, int Ci, int H, int W) {
  int K = Ci * 9;
  int total = H * W * K;
  int idx = blockIdx.x * blockDim.x + threadIdx.x;
  if (idx >= total) return;
  int m = idx / K, k = idx % K;
  int ci = k / 9, r = k % 9;
  int ky = r / 3, kx = r % 3;
  int y = m / W + ky - 1, x = m % W + kx - 1;
  float v = 0.f;
  if (y >= 0 && y < H && x >= 0 && x < W) v = in[((size_t)ci * H + y) * W + x];
  col[(size_t)m * K + k] = v;
}

// ---------------------------------------------------------------------------
// Small direct kernels
// ---------------------------------------------------------------------------
__global__ void conv1d_c0_kernel(const float* in, const float* w, const float* b, float* out) {
  int idx = blockIdx.x * blockDim.x + threadIdx.x;
  if (idx >= 32 * 512) return;
  int c = idx / 512, t = idx % 512;
  float acc = b[c];
  for (int i = 0; i < 18; i++)
    for (int j = 0; j < 3; j++) {
      int tt = t + j - 1;
      if (tt >= 0 && tt < 512) acc += w[(c * 18 + i) * 3 + j] * in[i * 512 + tt];
    }
  out[idx] = acc;
}

__global__ void mamba_dwconv_silu_kernel(const float* zx, const float* w, const float* b, float* out) {
  int idx = blockIdx.x * blockDim.x + threadIdx.x;
  if (idx >= 1280 * 32) return;
  int c = idx / 32, t = idx % 32;
  float acc = b[c];
  for (int j = 0; j < 4; j++) {
    int tt = t + j - 3;
    if (tt >= 0) acc += w[c * 4 + j] * zx[tt * 2320 + 1024 + c];
  }
  out[idx] = siluf(acc);
}

__global__ void mamba_dt_kernel(const float* zx, const float* dtb, float* dtm) {
  int idx = blockIdx.x * blockDim.x + threadIdx.x;
  if (idx >= 32 * 16) return;
  int t = idx / 16, h = idx % 16;
  dtm[idx] = softplusf(zx[t * 2320 + 2304 + h] + dtb[h]);
}

__global__ void mamba_scan_kernel(const float* xBCc, const float* dtm,
                                  const float* Alog, const float* Dv, float* ym) {
  int idx = blockIdx.x * blockDim.x + threadIdx.x;
  if (idx >= 1024) return;
  int head = idx >> 6;
  float A = -expf(Alog[head]);
  float Dd = Dv[head];
  const float* xrow = xBCc + (size_t)idx * 32;
  float h[128];
  for (int n = 0; n < 128; n++) h[n] = 0.f;
  for (int t = 0; t < 32; t++) {
    float dt = dtm[t * 16 + head];
    float dA = expf(dt * A);
    float xv = xrow[t];
    float du = dt * xv;
    float y = 0.f;
    for (int n = 0; n < 128; n++) {
      float Bv = xBCc[(size_t)(1024 + n) * 32 + t];
      float Cv = xBCc[(size_t)(1152 + n) * 32 + t];
      h[n] = h[n] * dA + du * Bv;
      y += h[n] * Cv;
    }
    ym[t * 1024 + idx] = y + Dd * xv;
  }
}

__global__ void mamba_gate_rms_kernel(const float* ym, const float* zx,
                                      const float* nw, float* tn) {
  int row = blockIdx.x, lane = threadIdx.x;
  float ss = 0.f;
  for (int i = lane; i < 1024; i += 32) {
    float z = zx[row * 2320 + i];
    float t = ym[row * 1024 + i] * siluf(z);
    tn[row * 1024 + i] = t;
    ss += t * t;
  }
  ss = wave_sum(ss);
  float scale = rsqrtf(ss / 1024.f + 1e-5f);
  for (int i = lane; i < 1024; i += 32)
    tn[row * 1024 + i] = tn[row * 1024 + i] * scale * nw[i];
}

__global__ void conv1d_c1_kernel(const float* in /*(32,512)*/, const float* w,
                                 const float* b, float* out /*(1024,32)*/) {
  int idx = blockIdx.x * blockDim.x + threadIdx.x;
  if (idx >= 1024 * 32) return;
  int o = idx / 32, t = idx % 32;
  float acc = b[o];
  for (int j = 0; j < 3; j++) {
    int tt = t + j - 1;
    if (tt < 0 || tt >= 32) continue;
    const float* ir = in + (size_t)tt * 512;
    const float* wr = w + (size_t)o * 1536 + j;
    float s = 0.f;
    for (int i = 0; i < 512; i++) s += ir[i] * wr[i * 3];
    acc += s;
  }
  out[idx] = acc;
}

__global__ void ln_rows_kernel(const float* in, const float* w, const float* b,
                               float* out, int C) {
  int row = blockIdx.x, lane = threadIdx.x;
  const float* r = in + (size_t)row * C;
  float s = 0.f, s2 = 0.f;
  for (int i = lane; i < C; i += 32) { float v = r[i]; s += v; s2 += v * v; }
  s = wave_sum(s); s2 = wave_sum(s2);
  float mu = s / C;
  float inv = rsqrtf(s2 / C - mu * mu + 1e-5f);
  float* o = out + (size_t)row * C;
  for (int i = lane; i < C; i += 32) o[i] = (r[i] - mu) * inv * w[i] + b[i];
}

// depthwise 3x3 + silu reading channels-last xz (pixel-major, ncol stride)
__global__ void dwconv3_silu_cl_kernel(const float* xz, int ncol, int res, int d,
                                       const float* w, const float* b, float* out) {
  int idx = blockIdx.x * blockDim.x + threadIdx.x;
  int total = d * res * res;
  if (idx >= total) return;
  int dd = idx / (res * res);
  int rem = idx % (res * res);
  int y = rem / res, x = rem % res;
  float acc = b[dd];
  for (int dy = 0; dy < 3; dy++)
    for (int dx = 0; dx < 3; dx++) {
      int yy = y + dy - 1, xx = x + dx - 1;
      if (yy < 0 || yy >= res || xx < 0 || xx >= res) continue;
      acc += w[dd * 9 + dy * 3 + dx] * xz[(size_t)(yy * res + xx) * ncol + dd];
    }
  out[idx] = siluf(acc);
}

__global__ void build_xs_kernel(const float* xin, float* xs, int d, int res) {
  int L = res * res;
  int idx = blockIdx.x * blockDim.x + threadIdx.x;
  if (idx >= d * L) return;
  int dd = idx / L, l = idx % L;
  size_t dL = (size_t)d * L;
  int i2 = (l % res) * res + l / res;
  int l3 = L - 1 - l;
  int i3 = (l3 % res) * res + l3 / res;
  const float* row = xin + (size_t)dd * L;
  float* o = xs + (size_t)dd * L + l;
  o[0]      = row[l];
  o[dL]     = row[i2];
  o[2 * dL] = row[L - 1 - l];
  o[3 * dL] = row[i3];
}

__global__ void dtproj_softplus_kernel(const float* dbl_k, const float* w_k,
                                       const float* b_k, float* out_k,
                                       int r, int d, int L) {
  int idx = blockIdx.x * blockDim.x + threadIdx.x;
  if (idx >= d * L) return;
  int dd = idx / L, l = idx % L;
  float acc = b_k[dd];
  for (int rr = 0; rr < r; rr++) acc += dbl_k[(size_t)rr * L + l] * w_k[dd * r + rr];
  out_k[idx] = softplusf(acc);
}

__global__ void selective_scan4_kernel(const float* xs, const float* dts,
                                       const float* dbl, const float* Alog,
                                       const float* Ds, float* ys,
                                       int d, int L, int cdim, int r) {
  int idx = blockIdx.x * blockDim.x + threadIdx.x;
  if (idx >= 4 * d) return;
  int k = idx / d, dd = idx % d;
  const float* u  = xs  + (size_t)idx * L;
  const float* dt = dts + (size_t)idx * L;
  const float* Bp = dbl + ((size_t)k * cdim + r) * L;
  const float* Cp = dbl + ((size_t)k * cdim + r + 16) * L;
  float A[16], h[16];
#pragma unroll
  for (int n = 0; n < 16; n++) {
    A[n] = -expf(Alog[(size_t)(k * d + dd) * 16 + n]);
    h[n] = 0.f;
  }
  float Dv = Ds[k * d + dd];
  float* yo = ys + (size_t)idx * L;
  for (int l = 0; l < L; l++) {
    float dtv = dt[l], uv = u[l];
    float du = dtv * uv;
    float y = 0.f;
#pragma unroll
    for (int n = 0; n < 16; n++) {
      float dA = expf(dtv * A[n]);
      h[n] = h[n] * dA + du * Bp[(size_t)n * L + l];
      y += h[n] * Cp[(size_t)n * L + l];
    }
    yo[l] = y + Dv * uv;
  }
}

__global__ void combine_dirs_kernel(const float* ys, float* ycomb, int d, int res) {
  int L = res * res;
  int idx = blockIdx.x * blockDim.x + threadIdx.x;
  if (idx >= L * d) return;
  int l = idx / d, dd = idx % d;
  size_t dL = (size_t)d * L;
  int i2 = (l % res) * res + l / res;
  int l3 = L - 1 - l;
  int i3 = (l3 % res) * res + l3 / res;
  const float* row = ys + (size_t)dd * L;
  ycomb[idx] = row[l] + row[dL + i2] + row[2 * dL + (L - 1 - l)] + row[3 * dL + i3];
}

__global__ void ln_gate_kernel(const float* ycomb, const float* w, const float* b,
                               const float* xz, int zoff, int ncol, float* out, int d) {
  int row = blockIdx.x, lane = threadIdx.x;
  const float* r = ycomb + (size_t)row * d;
  float s = 0.f, s2 = 0.f;
  for (int i = lane; i < d; i += 32) { float v = r[i]; s += v; s2 += v * v; }
  s = wave_sum(s); s2 = wave_sum(s2);
  float mu = s / d;
  float inv = rsqrtf(s2 / d - mu * mu + 1e-5f);
  for (int i = lane; i < d; i += 32) {
    float z = xz[(size_t)row * ncol + zoff + i];
    out[(size_t)row * d + i] = ((r[i] - mu) * inv * w[i] + b[i]) * siluf(z);
  }
}

__global__ void add_rows_kernel(const float* a, const float* b, float* o, int n) {
  int idx = blockIdx.x * blockDim.x + threadIdx.x;
  if (idx < n) o[idx] = a[idx] + b[idx];
}

__global__ void transpose_rc_kernel(const float* in, float* out, int R, int C) {
  int idx = blockIdx.x * blockDim.x + threadIdx.x;
  if (idx >= R * C) return;
  int r = idx / C, c = idx % C;
  out[(size_t)c * R + r] = in[idx];
}

__global__ void conv2d_g_kernel(const float* in, const float* w, const float* b,
                                float* out, int Ci, int Co, int H, int Wd,
                                int Ksz, int pad, int groups) {
  int idx = blockIdx.x * blockDim.x + threadIdx.x;
  int total = Co * H * Wd;
  if (idx >= total) return;
  int co = idx / (H * Wd);
  int rem = idx % (H * Wd);
  int y = rem / Wd, x = rem % Wd;
  int cig = Ci / groups, cog = Co / groups;
  int g = co / cog;
  float acc = b[co];
  for (int ci = 0; ci < cig; ci++) {
    int cin = g * cig + ci;
    for (int ky = 0; ky < Ksz; ky++) {
      int yy = y + ky - pad;
      if (yy < 0 || yy >= H) continue;
      for (int kx = 0; kx < Ksz; kx++) {
        int xx = x + kx - pad;
        if (xx < 0 || xx >= Wd) continue;
        acc += w[((size_t)(co * cig + ci) * Ksz + ky) * Ksz + kx] *
               in[((size_t)cin * H + yy) * Wd + xx];
      }
    }
  }
  out[idx] = acc;
}

__global__ void resize_bilinear_kernel(const float* in, float* out, int C, int Hi, int Ho) {
  int idx = blockIdx.x * blockDim.x + threadIdx.x;
  int total = C * Ho * Ho;
  if (idx >= total) return;
  int c = idx / (Ho * Ho);
  int rem = idx % (Ho * Ho);
  int y = rem / Ho, x = rem % Ho;
  float scale = (float)Hi / (float)Ho;
  float sy = (y + 0.5f) * scale - 0.5f;
  float sx = (x + 0.5f) * scale - 0.5f;
  int y0 = (int)floorf(sy); float fy = sy - y0;
  int x0 = (int)floorf(sx); float fx = sx - x0;
  int y1 = y0 + 1, x1 = x0 + 1;
  y0 = min(Hi - 1, max(0, y0)); y1 = min(Hi - 1, max(0, y1));
  x0 = min(Hi - 1, max(0, x0)); x1 = min(Hi - 1, max(0, x1));
  const float* p = in + (size_t)c * Hi * Hi;
  out[idx] = (1.f - fy) * ((1.f - fx) * p[y0 * Hi + x0] + fx * p[y0 * Hi + x1]) +
             fy * ((1.f - fx) * p[y1 * Hi + x0] + fx * p[y1 * Hi + x1]);
}

__global__ void final_out_kernel(const float* xv2, const float* proj2, float* out) {
  int idx = blockIdx.x * blockDim.x + threadIdx.x;
  if (idx >= 4096 * 512) return;
  int l = idx / 512, c = idx % 512;
  out[(size_t)c * 4096 + l] = xv2[idx] + proj2[idx];
}

// ---------------------------------------------------------------------------
extern "C" void kernel_launch(void* const* d_in, const int* in_sizes, int n_in,
                              void* d_out, int out_size, void* d_ws, size_t ws_size,
                              hipStream_t stream) {
  (void)in_sizes; (void)n_in; (void)out_size; (void)ws_size;
  const float* P[53];
  for (int i = 0; i < 53; i++) P[i] = (const float*)d_in[i];
  const float* latent = P[0];
  const float *c0_w = P[1], *c0_b = P[2];
  const float *m_in_w = P[3], *m_conv_w = P[4], *m_conv_b = P[5];
  const float *m_dt_bias = P[6], *m_A_log = P[7], *m_D = P[8];
  const float *m_norm_w = P[9], *m_out_w = P[10];
  const float *c1_w = P[11], *c1_b = P[12];
  const float *v1_ln_w = P[13], *v1_ln_b = P[14], *v1_in_w = P[15];
  const float *v1_conv_w = P[16], *v1_conv_b = P[17], *v1_xp_w = P[18];
  const float *v1_dt_w = P[19], *v1_dt_b = P[20], *v1_Alog = P[21], *v1_D = P[22];
  const float *v1_on_w = P[23], *v1_on_b = P[24], *v1_out_w = P[25];
  const float *dw1_w = P[26], *dw1_b = P[27], *pw1_w = P[28], *pw1_b = P[29];
  const float *dw2_w = P[30], *dw2_b = P[31], *pw2_w = P[32], *pw2_b = P[33];
  const float *cv1_w = P[34], *cv1_b = P[35], *cv2_w = P[36], *cv2_b = P[37];
  const float *fc_w = P[38], *fc_b = P[39];
  const float *v2_ln_w = P[40], *v2_ln_b = P[41], *v2_in_w = P[42];
  const float *v2_conv_w = P[43], *v2_conv_b = P[44], *v2_xp_w = P[45];
  const float *v2_dt_w = P[46], *v2_dt_b = P[47], *v2_Alog = P[48], *v2_D = P[49];
  const float *v2_on_w = P[50], *v2_on_b = P[51], *v2_out_w = P[52];

  float* W = (float*)d_ws;
  size_t off = 0;
  auto alloc = [&](size_t n) { float* p = W + off; off += n; return p; };
  auto g1 = [](int n) { return dim3((unsigned)((n + 255) / 256)); };
  auto gemm = [&](const float* A, const float* B, const float* bias, float* Cp,
                  int M, int N, int K, int aCol, int oCol) {
    dim3 grid((N + 127) / 128, (M + 63) / 64);
    gemm_wmma_kernel<<<grid, 256, 0, stream>>>(A, B, bias, Cp, M, N, K, aCol, oCol);
  };

  // persistent across both phases
  float* xv2 = alloc((size_t)4096 * 512);
  size_t mark = off;

  // ---- phase 1: conv1d + mamba2 + c1 + vssblock1 + conv stack -> xv2 ----
  float* c0o  = alloc(32 * 512);
  float* zx   = alloc(32 * 2320);
  float* xBCc = alloc(1280 * 32);
  float* dtm  = alloc(32 * 16);
  float* ym   = alloc(32 * 1024);
  float* tn   = alloc(32 * 1024);
  float* mo   = alloc(32 * 512);
  float* c1o  = alloc(1024 * 32);
  float* ln1  = alloc(1024 * 32);
  float* xz1  = alloc(1024 * 128);
  float* xin1 = alloc(64 * 1024);
  float* xs1  = alloc(4 * 64 * 1024);
  float* dbl1 = alloc(4 * 34 * 1024);
  float* dts1 = alloc(4 * 64 * 1024);
  float* ys1  = alloc(4 * 64 * 1024);
  float* yc1  = alloc(1024 * 64);
  float* yg1  = alloc(1024 * 64);
  float* pr1  = alloc(1024 * 32);
  float* vss1 = alloc(1024 * 32);
  float* tch  = alloc(32 * 1024);
  float* dw1o = alloc(32 * 1024);
  float* pw1o = alloc(64 * 1024);
  float* dw2o = alloc(64 * 1024);
  float* pw2o = alloc(128 * 1024);
  float* up   = alloc(128 * 4096);
  float* cv1o = alloc(128 * 4096);
  float* cv2o = alloc((size_t)256 * 4096);
  float* colb = alloc((size_t)4096 * 2304);  // im2col scratch (max K=2304)

  conv1d_c0_kernel<<<g1(32 * 512), 256, 0, stream>>>(latent, c0_w, c0_b, c0o);
  gemm(c0o, m_in_w, nullptr, zx, 32, 2320, 512, 0, 0);
  mamba_dwconv_silu_kernel<<<g1(1280 * 32), 256, 0, stream>>>(zx, m_conv_w, m_conv_b, xBCc);
  mamba_dt_kernel<<<g1(32 * 16), 256, 0, stream>>>(zx, m_dt_bias, dtm);
  mamba_scan_kernel<<<g1(1024), 256, 0, stream>>>(xBCc, dtm, m_A_log, m_D, ym);
  mamba_gate_rms_kernel<<<32, 32, 0, stream>>>(ym, zx, m_norm_w, tn);
  gemm(tn, m_out_w, nullptr, mo, 32, 512, 1024, 0, 0);
  conv1d_c1_kernel<<<g1(1024 * 32), 256, 0, stream>>>(mo, c1_w, c1_b, c1o);

  // vssblock1 (res=32, C=32, d=64, r=2, cdim=34)
  ln_rows_kernel<<<1024, 32, 0, stream>>>(c1o, v1_ln_w, v1_ln_b, ln1, 32);
  gemm(ln1, v1_in_w, nullptr, xz1, 1024, 128, 32, 0, 0);
  dwconv3_silu_cl_kernel<<<g1(64 * 1024), 256, 0, stream>>>(xz1, 128, 32, 64, v1_conv_w, v1_conv_b, xin1);
  build_xs_kernel<<<g1(64 * 1024), 256, 0, stream>>>(xin1, xs1, 64, 32);
  for (int k = 0; k < 4; k++) {
    gemm(xs1 + (size_t)k * 64 * 1024, v1_xp_w + (size_t)k * 34 * 64, nullptr,
         dbl1 + (size_t)k * 34 * 1024, 1024, 34, 64, 1, 1);
    dtproj_softplus_kernel<<<g1(64 * 1024), 256, 0, stream>>>(
        dbl1 + (size_t)k * 34 * 1024, v1_dt_w + (size_t)k * 64 * 2,
        v1_dt_b + (size_t)k * 64, dts1 + (size_t)k * 64 * 1024, 2, 64, 1024);
  }
  selective_scan4_kernel<<<g1(4 * 64), 256, 0, stream>>>(xs1, dts1, dbl1, v1_Alog, v1_D, ys1, 64, 1024, 34, 2);
  combine_dirs_kernel<<<g1(1024 * 64), 256, 0, stream>>>(ys1, yc1, 64, 32);
  ln_gate_kernel<<<1024, 32, 0, stream>>>(yc1, v1_on_w, v1_on_b, xz1, 64, 128, yg1, 64);
  gemm(yg1, v1_out_w, nullptr, pr1, 1024, 32, 64, 0, 0);
  add_rows_kernel<<<g1(1024 * 32), 256, 0, stream>>>(c1o, pr1, vss1, 1024 * 32);

  // conv stack
  transpose_rc_kernel<<<g1(1024 * 32), 256, 0, stream>>>(vss1, tch, 1024, 32);
  conv2d_g_kernel<<<g1(32 * 1024), 256, 0, stream>>>(tch, dw1_w, dw1_b, dw1o, 32, 32, 32, 32, 3, 1, 32);
  gemm(dw1o, pw1_w, pw1_b, pw1o, 1024, 64, 32, 1, 1);
  conv2d_g_kernel<<<g1(64 * 1024), 256, 0, stream>>>(pw1o, dw2_w, dw2_b, dw2o, 64, 64, 32, 32, 3, 1, 64);
  gemm(dw2o, pw2_w, pw2_b, pw2o, 1024, 128, 64, 1, 1);
  resize_bilinear_kernel<<<g1(128 * 4096), 256, 0, stream>>>(pw2o, up, 128, 32, 64);
  // cv1 3x3 as im2col + WMMA GEMM (M=4096, N=128, K=1152) -> row-major (px, ch)
  im2col3x3_kernel<<<g1(4096 * 1152), 256, 0, stream>>>(up, colb, 128, 64, 64);
  gemm(colb, cv1_w, cv1_b, cv1o, 4096, 128, 1152, 0, 0);
  // cv2 1x1: A = cv1o row-major (px, 128) -> NCHW col-major out (256, 4096)
  gemm(cv1o, cv2_w, cv2_b, cv2o, 4096, 256, 128, 0, 1);
  // fc 3x3 as im2col + WMMA GEMM (M=4096, N=512, K=2304); row-major output
  // (pixel, channel) == (L, C) layout required by vssblock2 -> write xv2 directly
  im2col3x3_kernel<<<g1(4096 * 2304), 256, 0, stream>>>(cv2o, colb, 256, 64, 64);
  gemm(colb, fc_w, fc_b, xv2, 4096, 512, 2304, 0, 0);

  // ---- phase 2: vssblock2 (res=64, C=512, d=1024, r=32, cdim=64) ----
  off = mark;  // reuse phase-1 scratch
  float* ln2  = alloc((size_t)4096 * 512);
  float* xz2  = alloc((size_t)4096 * 2048);
  float* xin2 = alloc((size_t)1024 * 4096);
  float* xs2  = alloc((size_t)4 * 1024 * 4096);
  float* dbl2 = alloc((size_t)4 * 64 * 4096);
  float* dts2 = alloc((size_t)4 * 1024 * 4096);
  float* ys2  = alloc((size_t)4 * 1024 * 4096);
  float* yg2  = alloc((size_t)4096 * 1024);
  float* yc2  = xin2;  // xin2 dead after build_xs
  float* pr2  = ln2;   // ln2 dead after xz gemm

  ln_rows_kernel<<<4096, 32, 0, stream>>>(xv2, v2_ln_w, v2_ln_b, ln2, 512);
  gemm(ln2, v2_in_w, nullptr, xz2, 4096, 2048, 512, 0, 0);
  dwconv3_silu_cl_kernel<<<g1(1024 * 4096), 256, 0, stream>>>(xz2, 2048, 64, 1024, v2_conv_w, v2_conv_b, xin2);
  build_xs_kernel<<<g1(1024 * 4096), 256, 0, stream>>>(xin2, xs2, 1024, 64);
  for (int k = 0; k < 4; k++) {
    gemm(xs2 + (size_t)k * 1024 * 4096, v2_xp_w + (size_t)k * 64 * 1024, nullptr,
         dbl2 + (size_t)k * 64 * 4096, 4096, 64, 1024, 1, 1);
    dtproj_softplus_kernel<<<g1(1024 * 4096), 256, 0, stream>>>(
        dbl2 + (size_t)k * 64 * 4096, v2_dt_w + (size_t)k * 1024 * 32,
        v2_dt_b + (size_t)k * 1024, dts2 + (size_t)k * 1024 * 4096, 32, 1024, 4096);
  }
  selective_scan4_kernel<<<g1(4 * 1024), 256, 0, stream>>>(xs2, dts2, dbl2, v2_Alog, v2_D, ys2, 1024, 4096, 64, 32);
  combine_dirs_kernel<<<g1(4096 * 1024), 256, 0, stream>>>(ys2, yc2, 1024, 64);
  ln_gate_kernel<<<4096, 32, 0, stream>>>(yc2, v2_on_w, v2_on_b, xz2, 1024, 2048, yg2, 1024);
  gemm(yg2, v2_out_w, nullptr, pr2, 4096, 512, 1024, 0, 0);
  final_out_kernel<<<g1(4096 * 512), 256, 0, stream>>>(xv2, pr2, (float*)d_out);
}
